// RegimeMoE_22419729285678
// MI455X (gfx1250) — matrix operand order repaired and checked by reading, not compiled, not run on hardware
//
#include <hip/hip_runtime.h>
#include <hip/hip_bf16.h>
#include <math.h>

typedef __attribute__((ext_vector_type(16))) __bf16 v16bf;
typedef __attribute__((ext_vector_type(8)))  __bf16 v8bf;
typedef __attribute__((ext_vector_type(8)))  float  v8f;
typedef unsigned int v4u  __attribute__((ext_vector_type(4)));
typedef int          v8i_t __attribute__((ext_vector_type(8)));
typedef int          v4i_t __attribute__((ext_vector_type(4)));

#define B_TOTAL   131072
#define D_DIM     256
#define K_EXP     8
#define H_DIM     128
#define TILE_ROWS 128
#define NWAVES    8

// padded LDS pitches (elements); rows stay 16B-aligned, banks spread
#define ZB_PITCH 264
#define W1_PITCH 264
#define W2_PITCH 136
#define H1_PITCH 136

// LDS layout (byte offsets into one shared block, all 16B aligned)
#define ZB_OFF     0                      // 128*264*2          = 67584
#define W1T_OFF    67584                  // 2 buf * 128*264*2  = 135168
#define W1T_BUFSZ  67584
#define W2T_OFF    202752                 // 2 buf * 128*136*2  = 69632
#define W2T_BUFSZ  34816
#define H1S_OFF    272384                 // 8 waves * 16*136*2 = 34816
#define H1S_WAVESZ 4352
#define BIAS1_OFF  307200
#define BIAS2_OFF  307712
#define W3S_OFF    308224
#define B3S_OFF    308736
#define SMEM_BYTES 308752

__device__ __forceinline__ float gelu_exact(float x) {
    return 0.5f * x * (1.0f + erff(x * 0.70710678118654752f));
}

// A-fragment (16x32 bf16): lane=row; elems 0-7 at k=half*8, elems 8-15 at k=16+half*8
__device__ __forceinline__ v16bf load_a_frag(const __bf16* p) {
    union { v16bf v; v8bf h[2]; } u;
    u.h[0] = *(const v8bf*)(p);
    u.h[1] = *(const v8bf*)(p + 16);
    return u.v;
}
// B-fragment (32x16 bf16): lane=col; elems 0-7 at k=half*8, elems 8-15 at k=16+half*8
__device__ __forceinline__ v16bf load_b_frag(const __bf16* p) {
    union { v16bf v; v8bf h[2]; } u;
    u.h[0] = *(const v8bf*)(p);
    u.h[1] = *(const v8bf*)(p + 16);
    return u.v;
}

// ---- Tensor Data Mover: 2-D bf16 tile (rows x cols), row-major, LDS row padding ----
// pad_icode: DWORDs-per-row code (0=2,1=4,...,6=128); pad_acode: pad DWORDs-1
__device__ __forceinline__ void tdm_load_2d(unsigned lds_addr, const void* gaddr,
                                            unsigned cols, unsigned rows,
                                            unsigned pad_icode, unsigned pad_acode) {
    unsigned long long ga = (unsigned long long)gaddr;
    v4u g0;
    g0.x = 1u;                                             // count=1, user mode
    g0.y = lds_addr;                                       // lds byte address
    g0.z = (unsigned)(ga & 0xFFFFFFFFu);                   // global_addr[31:0]
    g0.w = (unsigned)((ga >> 32) & 0x1FFFFFFu) | (2u << 30); // addr[56:32] | type=2
    v8i_t g1;
    g1[0] = (int)((1u << 16)            // data_size = 1 -> 2 bytes
                | (1u << 20)            // pad_enable
                | (pad_icode << 22)
                | (pad_acode << 25));
    g1[1] = (int)((cols & 0xFFFFu) << 16);                         // tensor_dim0 lo16
    g1[2] = (int)(((cols >> 16) & 0xFFFFu) | ((rows & 0xFFFFu) << 16)); // dim0 hi | dim1 lo
    g1[3] = (int)(((rows >> 16) & 0xFFFFu) | ((cols & 0xFFFFu) << 16)); // dim1 hi | tile_dim0
    g1[4] = (int)(rows & 0xFFFFu);                                 // tile_dim1 (tile_dim2=0)
    g1[5] = (int)cols;                                             // tensor_dim0_stride lo32
    g1[6] = 0;
    g1[7] = 0;
    v4i_t g2 = {0, 0, 0, 0};
    v4i_t g3 = {0, 0, 0, 0};
    v8i_t g4 = {0, 0, 0, 0, 0, 0, 0, 0};
    __builtin_amdgcn_tensor_load_to_lds(g0, g1, g2, g3, g4, 0);
}

// ---- prologue kernel: W1 -> bf16 W1^T[k][h][d], W2 -> bf16 W2^T[k][g][h] ----
__global__ void convert_weights_kernel(const float* __restrict__ W1,
                                       const float* __restrict__ W2,
                                       __bf16* __restrict__ w1t,
                                       __bf16* __restrict__ w2t) {
    int tid = blockIdx.x * 256 + threadIdx.x;
    if (tid < K_EXP * H_DIM * D_DIM) {          // 262144
        int d = tid & 255, h = (tid >> 8) & 127, k = tid >> 15;
        w1t[tid] = (__bf16)W1[((size_t)k * D_DIM + d) * H_DIM + h];
    }
    if (tid < K_EXP * H_DIM * H_DIM) {          // 131072
        int h = tid & 127, g = (tid >> 7) & 127, k = tid >> 14;
        w2t[tid] = (__bf16)W2[((size_t)k * H_DIM + h) * H_DIM + g];
    }
}

__global__ __launch_bounds__(256, 1) void regime_moe_kernel(
    const float* __restrict__ z,  const float* __restrict__ rp,
    const __bf16* __restrict__ W1bf, const __bf16* __restrict__ W2bf,
    const float* __restrict__ b1, const float* __restrict__ b2,
    const float* __restrict__ W3, const float* __restrict__ b3,
    float* __restrict__ out)
{
    __shared__ __align__(16) unsigned char smem[SMEM_BYTES];
    __bf16* zb    = (__bf16*)(smem + ZB_OFF);
    float*  bias1 = (float*)(smem + BIAS1_OFF);
    float*  bias2 = (float*)(smem + BIAS2_OFF);
    float*  w3s   = (float*)(smem + W3S_OFF);
    float*  b3s   = (float*)(smem + B3S_OFF);
    const unsigned lds_base = (unsigned)(size_t)(void*)smem;  // LDS aperture: addr[31:0] = offset

    const int t    = threadIdx.x;
    const int wv   = t >> 5;
    const int lane = t & 31;
    const int half = lane >> 4;
    const int lp   = lane & 15;
    const int rowTile = blockIdx.x * TILE_ROWS;

    // prologue: async-DMA expert 0 weights into buffer 0 while we stage z
    if (wv == 0) {
        tdm_load_2d(lds_base + W1T_OFF, W1bf, D_DIM, H_DIM, /*128 DW rows*/6, /*4 DW pad*/3);
        tdm_load_2d(lds_base + W2T_OFF, W2bf, H_DIM, H_DIM, /*64 DW rows*/5, /*4 DW pad*/3);
    }
    for (int i = 0; i < TILE_ROWS; ++i)                       // z tile -> bf16 (coalesced)
        zb[i * ZB_PITCH + t] = (__bf16)z[(size_t)(rowTile + i) * D_DIM + t];
    if (t < H_DIM) { bias1[t] = b1[t]; bias2[t] = b2[t]; w3s[t] = W3[t]; }
    if (t == 0) *b3s = b3[0];
    if (wv == 0) __builtin_amdgcn_s_wait_tensorcnt(0);
    __syncthreads();

    v8f vzero;
    #pragma unroll
    for (int i = 0; i < 8; ++i) vzero[i] = 0.f;

    float bl[8];
    #pragma unroll
    for (int j = 0; j < 8; ++j) bl[j] = 0.f;

    const int rowBaseHalf = rowTile + wv * 16 + 8 * half;     // rows rowBaseHalf+j

    for (int k = 0; k < K_EXP; ++k) {
        const int cur = k & 1, nxt = cur ^ 1;
        // kick off next expert's weight DMA into the other buffer (overlaps compute)
        if (wv == 0 && k + 1 < K_EXP) {
            tdm_load_2d(lds_base + W1T_OFF + nxt * W1T_BUFSZ,
                        W1bf + (size_t)(k + 1) * H_DIM * D_DIM, D_DIM, H_DIM, 6, 3);
            tdm_load_2d(lds_base + W2T_OFF + nxt * W2T_BUFSZ,
                        W2bf + (size_t)(k + 1) * H_DIM * H_DIM, H_DIM, H_DIM, 5, 3);
        }
        const __bf16* w1t = (const __bf16*)(smem + W1T_OFF + cur * W1T_BUFSZ);
        const __bf16* w2t = (const __bf16*)(smem + W2T_OFF + cur * W2T_BUFSZ);
        __bf16* h1w = (__bf16*)(smem + H1S_OFF + wv * H1S_WAVESZ);

        // ---- layer 1: (16x256) @ (256x128) per wave ----
        v8f acc[8];
        #pragma unroll
        for (int n = 0; n < 8; ++n) acc[n] = vzero;
        const __bf16* aRow = &zb[(wv * 16 + lp) * ZB_PITCH];
        #pragma unroll
        for (int kk = 0; kk < 8; ++kk) {
            v16bf a = load_a_frag(aRow + kk * 32 + half * 8);
            #pragma unroll
            for (int n = 0; n < 8; ++n) {
                v16bf bf = load_b_frag(&w1t[(n * 16 + lp) * W1_PITCH + kk * 32 + half * 8]);
                acc[n] = __builtin_amdgcn_wmma_f32_16x16x32_bf16(
                    false, a, false, bf, (short)0, acc[n], false, false);
            }
        }
        #pragma unroll
        for (int n = 0; n < 8; ++n) {
            int c = n * 16 + lp;
            float bb = bias1[c];
            #pragma unroll
            for (int j = 0; j < 8; ++j) {
                float g = gelu_exact(acc[n][j] + bb);
                h1w[(j + 8 * half) * H1_PITCH + c] = (__bf16)g;
            }
        }

        // ---- layer 2: (16x128) @ (128x128) per wave ----
        v8f acc2[8];
        #pragma unroll
        for (int n = 0; n < 8; ++n) acc2[n] = vzero;
        const __bf16* aRow2 = &h1w[lp * H1_PITCH];
        #pragma unroll
        for (int kk = 0; kk < 4; ++kk) {
            v16bf a = load_a_frag(aRow2 + kk * 32 + half * 8);
            #pragma unroll
            for (int n = 0; n < 8; ++n) {
                v16bf bf = load_b_frag(&w2t[(n * 16 + lp) * W2_PITCH + kk * 32 + half * 8]);
                acc2[n] = __builtin_amdgcn_wmma_f32_16x16x32_bf16(
                    false, a, false, bf, (short)0, acc2[n], false, false);
            }
        }

        // ---- layer 3: bias + GELU + dot(W3) ----
        float pacc[8];
        #pragma unroll
        for (int j = 0; j < 8; ++j) pacc[j] = 0.f;
        #pragma unroll
        for (int n = 0; n < 8; ++n) {
            int c = n * 16 + lp;
            float bb = bias2[c], w3c = w3s[c];
            #pragma unroll
            for (int j = 0; j < 8; ++j)
                pacc[j] += gelu_exact(acc2[n][j] + bb) * w3c;
        }
        #pragma unroll
        for (int j = 0; j < 8; ++j) {
            pacc[j] += __shfl_xor(pacc[j], 8, 32);
            pacc[j] += __shfl_xor(pacc[j], 4, 32);
            pacc[j] += __shfl_xor(pacc[j], 2, 32);
            pacc[j] += __shfl_xor(pacc[j], 1, 32);
        }
        float bconst = *b3s;
        if (lp == 0) {
            #pragma unroll
            for (int j = 0; j < 8; ++j) {
                int row = rowBaseHalf + j;
                float pred = pacc[j] + bconst;
                out[B_TOTAL + (size_t)row * K_EXP + k] = pred;   // expert_preds
                bl[j] += rp[(size_t)row * K_EXP + k] * pred;     // blend
            }
        }

        __syncthreads();   // all waves done with cur buffers + bias arrays
        if (k + 1 < K_EXP) {
            if (t < H_DIM) {
                bias1[t] = b1[(k + 1) * H_DIM + t];
                bias2[t] = b2[(k + 1) * H_DIM + t];
                w3s[t]   = W3[(k + 1) * H_DIM + t];
            }
            if (t == 0) *b3s = b3[k + 1];
            if (wv == 0) __builtin_amdgcn_s_wait_tensorcnt(0);   // next weights landed
        }
        __syncthreads();   // publish next buffers/biases
    }

    if (lp == 0) {
        #pragma unroll
        for (int j = 0; j < 8; ++j)
            out[rowBaseHalf + j] = bl[j];                        // blended
    }
}

extern "C" void kernel_launch(void* const* d_in, const int* in_sizes, int n_in,
                              void* d_out, int out_size, void* d_ws, size_t ws_size,
                              hipStream_t stream) {
    const float* z  = (const float*)d_in[0];
    const float* rp = (const float*)d_in[1];
    const float* W1 = (const float*)d_in[2];
    const float* b1 = (const float*)d_in[3];
    const float* W2 = (const float*)d_in[4];
    const float* b2 = (const float*)d_in[5];
    const float* W3 = (const float*)d_in[6];
    const float* b3 = (const float*)d_in[7];
    float* out = (float*)d_out;

    // scratch: bf16 transposed weights (512KB + 256KB)
    __bf16* w1t = (__bf16*)d_ws;
    __bf16* w2t = (__bf16*)((char*)d_ws + (size_t)K_EXP * H_DIM * D_DIM * 2);

    hipLaunchKernelGGL(convert_weights_kernel, dim3(1024), dim3(256), 0, stream,
                       W1, W2, w1t, w2t);
    hipLaunchKernelGGL(regime_moe_kernel, dim3(B_TOTAL / TILE_ROWS), dim3(256), 0, stream,
                       z, rp, w1t, w2t, b1, b2, W3, b3, out);
}
